// Markonv_stride_3015067041991
// MI455X (gfx1250) — compile-verified, matrix operand order checked by reading, not run on hardware
//
#include <hip/hip_runtime.h>
#include <hip/hip_bf16.h>

// ---- Markonv as implicit GEMM on CDNA5 WMMA (f16 in, f32 acc) ----
// out[b,n,p] = sum_{l,i,j} x[b,i,p*s+l] * x[b,j,p*s+l+1] * W[l,i,j,n]
// Reduction dim R = K*C*C = 320 = 10 blocks of 32 -> v_wmma_f32_16x16x32_f16.

typedef __attribute__((ext_vector_type(16))) _Float16 v16h;
typedef __attribute__((ext_vector_type(8)))  _Float16 v8h;
typedef __attribute__((ext_vector_type(8)))  float    v8f;

// Problem constants (match reference harness)
constexpr int Bc = 128;    // batch
constexpr int Cc = 4;      // channels
constexpr int Lc = 10000;  // sequence length
constexpr int Kc = 20;     // kernel taps
constexpr int Nc = 128;    // filters
constexpr int KB = (Kc * Cc * Cc) / 32;   // 10 k-blocks of 32
constexpr int M_TILE  = 128;              // positions per LDS fill
constexpr int PCHUNK  = 1280;             // positions per workgroup
constexpr int NCHUNK  = (Lc - Kc + PCHUNK - 1) / PCHUNK;  // 8 (for s=1)
constexpr int PAIR_T  = 160;              // LDS t extent (>= 127*s + 21 for s=1)

// ---------------------------------------------------------------------------
// Prep kernel: reorder W[l,i,j,n] (f32) into per-lane WMMA B fragments (f16).
// Layout: Bmat[((w*KB + kb)*32 + lane)*16 + h]
//   lane: n_local = lane&15, kkSel = lane>>4
//   h<8  -> kk = kkSel*8 + h        (VGPR 0..3 halves)
//   h>=8 -> kk = 16 + kkSel*8 + h-8 (VGPR 4..7 halves)
//   r = kb*32 + kk ; l = r>>4 ; c2 = r&15 ; i = c2>>2 ; j = c2&3
// ---------------------------------------------------------------------------
__global__ __launch_bounds__(256) void markonv_prepB(const float* __restrict__ W,
                                                     _Float16* __restrict__ Bmat) {
  int idx = blockIdx.x * blockDim.x + threadIdx.x;     // one thread per (w,kb,lane)
  if (idx >= (Nc / 16) * KB * 32) return;
  int lane  = idx & 31;
  int kb    = (idx >> 5) % KB;
  int w     = idx / (32 * KB);
  int nl    = lane & 15;
  int kkSel = lane >> 4;

  _Float16 vals[16];
#pragma unroll
  for (int h = 0; h < 16; ++h) {
    int kk = (h < 8) ? (kkSel * 8 + h) : (16 + kkSel * 8 + (h - 8));
    int r  = kb * 32 + kk;
    int l  = r >> 4;
    int c2 = r & 15;
    int i  = c2 >> 2;
    int j  = c2 & 3;
    int n  = w * 16 + nl;
    vals[h] = (_Float16)W[(((size_t)l * Cc + i) * Cc + j) * Nc + n];
  }
  _Float16* dst = Bmat + (size_t)idx * 16;
#pragma unroll
  for (int h = 0; h < 16; ++h) dst[h] = vals[h];
}

// ---------------------------------------------------------------------------
// Fill one pair-feature LDS buffer: pairBuf[c2half][t][q] for t in [0, TFILL).
// pair[t][c2=(i,j)] = x[b,i,t0+t] * x[b,j,t0+t+1], stored as f16.
// ---------------------------------------------------------------------------
__device__ __forceinline__ void fill_pair(_Float16 (*pairBuf)[PAIR_T][8],
                                          const float* __restrict__ xb,
                                          int p0, int s, int tid) {
  const int TFILL = (M_TILE - 1) * s + Kc;  // t_local in [0, TFILL)
  for (int t = tid; t < TFILL; t += 256) {
    long tg = (long)p0 * s + t;
    if (tg > (long)Lc - 2) tg = (long)Lc - 2;  // clamp (tail garbage unused)
    float x0[4], x1[4];
#pragma unroll
    for (int c = 0; c < 4; ++c) {
      x0[c] = xb[(size_t)c * Lc + tg];
      x1[c] = xb[(size_t)c * Lc + tg + 1];
    }
    // gfx1250 prefetch two tiles ahead (global_prefetch_b8)
    __builtin_prefetch(&xb[tg + 2L * M_TILE * s], 0, 3);

    v8h r0, r1;
#pragma unroll
    for (int j = 0; j < 4; ++j) {
      r0[j]     = (_Float16)(x0[0] * x1[j]);  // c2 = 0*4+j
      r0[4 + j] = (_Float16)(x0[1] * x1[j]);  // c2 = 1*4+j
      r1[j]     = (_Float16)(x0[2] * x1[j]);  // c2 = 2*4+j
      r1[4 + j] = (_Float16)(x0[3] * x1[j]);  // c2 = 3*4+j
    }
    *(v8h*)&pairBuf[0][t][0] = r0;  // ds_store_b128
    *(v8h*)&pairBuf[1][t][0] = r1;  // ds_store_b128
  }
}

// ---------------------------------------------------------------------------
// Main kernel: 8 waves per block; wave w owns filters [16w, 16w+16).
// Block covers one batch b and PCHUNK output positions, M_TILE at a time,
// with double-buffered pair-feature LDS (fill tile i+1 before WMMA of tile i).
// ---------------------------------------------------------------------------
__global__ __launch_bounds__(256) void markonv_wmma(const float* __restrict__ x,
                                                    const _Float16* __restrict__ Bmat,
                                                    const int* __restrict__ stride_ptr,
                                                    float* __restrict__ out) {
  __shared__ __attribute__((aligned(32))) _Float16 pairA[2][2][PAIR_T][8];

  int s = stride_ptr[0];
  if (s < 1) s = 1;
  const int P = (Lc - 1 - Kc) / s + 1;   // output length
  const bool vecOK = ((P & 3) == 0);     // f32x4-aligned output rows

  const int b     = blockIdx.x % Bc;
  const int chunk = blockIdx.x / Bc;
  const int p0c   = chunk * PCHUNK;

  const int tid   = threadIdx.x;
  const int wave  = tid >> 5;
  const int lane  = tid & 31;
  const int nl    = lane & 15;   // A row / D column within tile
  const int kkSel = lane >> 4;   // which half of kk this lane holds

  // ---- Load this wave's 10 B fragments into registers (reused all chunk) ----
  v16h bfrag[KB];
  {
    const _Float16* bp = Bmat + ((size_t)(wave * KB) * 32 + lane) * 16;
#pragma unroll
    for (int kb = 0; kb < KB; ++kb)
      bfrag[kb] = *(const v16h*)(bp + (size_t)kb * 32 * 16);  // 32B aligned
  }

  const float* xb = x + (size_t)b * Cc * Lc;
  float* const outb = out + (size_t)b * Nc * (size_t)P + ((size_t)wave * 16 + nl) * (size_t)P;

  // Prologue: fill buffer 0 for the first tile.
  fill_pair(pairA[0], xb, p0c, s, tid);
  __syncthreads();

  int buf = 0;
  for (int p0 = p0c; p0 < p0c + PCHUNK; p0 += M_TILE) {
    // Issue next tile's fill first: its global loads + ds_stores retire
    // underneath the 80 WMMAs below (double buffering, one barrier/tile).
    const int nxt = buf ^ 1;
    if (p0 + M_TILE < p0c + PCHUNK)
      fill_pair(pairA[nxt], xb, p0 + M_TILE, s, tid);

    // ---------------- 8 x (16 pos x 16 filt) WMMA tiles per wave ----------------
#pragma unroll 1
    for (int mt = 0; mt < M_TILE / 16; ++mt) {
      v8f acc = {};
      const int tbase = (mt * 16 + nl) * s;  // t_local for this lane's A row
#pragma unroll
      for (int kb = 0; kb < KB; ++kb) {
        const _Float16* ap = &pairA[buf][kkSel][tbase + 2 * kb][0];
        union { v16h v; v8h h[2]; } au;
        au.h[0] = *(const v8h*)(ap);       // ds_load_b128: l = 2kb
        au.h[1] = *(const v8h*)(ap + 8);   // ds_load_b128: l = 2kb+1
        acc = __builtin_amdgcn_wmma_f32_16x16x32_f16(
            /*neg_a=*/false, au.v, /*neg_b=*/false, bfrag[kb],
            /*c_mod=*/(short)0, acc, /*reuse_a=*/false, /*reuse_b=*/false);
      }
      // D layout: lane holds 8 consecutive p values (M = kkSel*8 + r) for n = nl
      const int pbase = p0 + mt * 16 + kkSel * 8;
      float* op = outb + pbase;
      union { v8f v; float4 q[2]; float f[8]; } r;
      r.v = acc;
      if (vecOK && pbase + 8 <= P) {
        *(float4*)(op)     = r.q[0];   // global_store_b128
        *(float4*)(op + 4) = r.q[1];   // global_store_b128
      } else {
#pragma unroll
        for (int j = 0; j < 8; ++j)
          if (pbase + j < P) op[j] = r.f[j];
      }
    }

    __syncthreads();  // next buffer's fill complete; current buffer free for reuse
    buf = nxt;
  }
}

// ---------------------------------------------------------------------------
extern "C" void kernel_launch(void* const* d_in, const int* in_sizes, int n_in,
                              void* d_out, int out_size, void* d_ws, size_t ws_size,
                              hipStream_t stream) {
  const float* x      = (const float*)d_in[0];
  const float* W      = (const float*)d_in[1];
  const int*   stride = (const int*)d_in[2];
  float*       out    = (float*)d_out;
  _Float16*    Bmat   = (_Float16*)d_ws;  // (Nc/16)*KB*32*16 halves = 80 KB

  const int prepThreads = (Nc / 16) * KB * 32;  // 2560
  markonv_prepB<<<(prepThreads + 255) / 256, 256, 0, stream>>>(W, Bmat);
  markonv_wmma<<<dim3(Bc * NCHUNK), dim3(256), 0, stream>>>(x, Bmat, stride, out);
}
